// WindowedAttention_27754078667085
// MI455X (gfx1250) — matrix operand order repaired and checked by reading, not compiled, not run on hardware
//
#include <hip/hip_runtime.h>

typedef unsigned short u16;
typedef unsigned int   u32;
typedef __bf16 v16bf __attribute__((ext_vector_type(16)));
typedef float  v8f   __attribute__((ext_vector_type(8)));

union Frag { u32 u[8]; v16bf v; };

#define C_DIM 768
#define C3    2304
#define LP    288      // padded token rows per window (multiple of 32)
#define NWIN  100      // B_ = 4 * 25
#define NHEAD 12

__device__ __forceinline__ u16 f2bf(float f) {
    union { float f; u32 u; } x; x.f = f;
    u32 r = (x.u + 0x7FFFu + ((x.u >> 16) & 1u)) >> 16;
    return (u16)r;
}
__device__ __forceinline__ float bf2f(u16 h) {
    union { u32 u; float f; } x; x.u = ((u32)h) << 16; return x.f;
}
__device__ __forceinline__ v8f wmma_bf(const Frag& A, const Frag& B, v8f C) {
    return __builtin_amdgcn_wmma_f32_16x16x32_bf16(false, A.v, false, B.v, (short)0, C, false, false);
}
// ISA 16-bit A-matrix 16x32 layout: lane m=L&15, half=L>>4
__device__ __forceinline__ int a_kidx(int v, int half) { return ((v >= 4) ? 16 : 0) + half * 8 + (v & 3) * 2; }
// ISA 16-bit B-matrix 32x16 layout: lane n=L&15, half=L>>4
__device__ __forceinline__ int b_kidx(int v, int half) { return ((v >= 4) ? 16 : 0) + half * 8 + (v & 3) * 2; }

__device__ __forceinline__ void async_lds_b128(unsigned lds_off, const void* g) {
    asm volatile("global_load_async_to_lds_b128 %0, %1, off"
                 :: "v"(lds_off), "v"((unsigned long long)(size_t)g)
                 : "memory");
}
__device__ __forceinline__ void wait_async0() {
    asm volatile("s_wait_asynccnt 0x0" ::: "memory");
}
__device__ __forceinline__ void wait_ds0() {
    asm volatile("s_wait_dscnt 0x0" ::: "memory");
}

// ---------------------------------------------------------------- kernel 0
__global__ void k_convert_weights(const float* __restrict__ w_in, const float* __restrict__ w_out,
                                  u16* __restrict__ win_bf, u16* __restrict__ wout_bf) {
    int i = blockIdx.x * 256 + threadIdx.x;
    const int NW_IN = C3 * C_DIM;
    if (i < NW_IN) win_bf[i] = f2bf(w_in[i]);
    else {
        int j = i - NW_IN;
        if (j < C_DIM * C_DIM) wout_bf[j] = f2bf(w_out[j]);
    }
}

// ---------------------------------------------------------------- kernel 1
// Build tok (NWIN x LP x C) bf16 (window gather + padded-zero + query replicate)
// and additive key mask (NWIN x LP) f32.
__global__ void k_pack(const float* __restrict__ x, const float* __restrict__ q,
                       const int* __restrict__ q_mask,
                       u16* __restrict__ tok, float* __restrict__ maskadd) {
    long long idx = (long long)blockIdx.x * 256 + threadIdx.x;
    if (idx >= (long long)NWIN * LP * C_DIM) return;
    int c = (int)(idx % C_DIM);
    long long r = idx / C_DIM;
    int l  = (int)(r % LP);
    int b_ = (int)(r / LP);
    float val = 0.0f;
    if (l < 196) {                 // window tokens: batch-major window order b_ = b*25 + wy*5 + wx
        int b = b_ / 25, w = b_ % 25;
        int wy = w / 5, wx = w % 5;
        int ty = l / 14, tx = l % 14;
        int y = wy * 14 + ty, xx = wx * 14 + tx;
        if (y < 64 && xx < 64) val = x[((size_t)b * 4096 + y * 64 + xx) * C_DIM + c];
    } else if (l < 260) {          // query tokens: repeat-major replicate -> batch = b_ % 4
        int j = l - 196, bq = b_ & 3;
        val = q[((size_t)bq * 64 + j) * C_DIM + c];
    }
    tok[(size_t)r * C_DIM + c] = f2bf(val);
    if (c == 0) {
        float m = 0.0f;
        if (l >= 260) m = -1e30f;
        else if (l >= 196) { if (q_mask[(b_ & 3) * 64 + (l - 196)] == 0) m = -1e30f; }
        maskadd[b_ * LP + l] = m;
    }
}

// ---------------------------------------------------------------- kernel 2
// QKV = tok @ W_in^T   -> (NWIN x LP x 2304) bf16.  A staged via async-to-LDS.
__global__ void k_qkv_gemm(const u16* __restrict__ tok, const u16* __restrict__ win_bf,
                           u16* __restrict__ qkv) {
    const int nbase = blockIdx.x * 64;
    const int m0    = blockIdx.y * 32;
    const int b_    = blockIdx.z;
    const int tid = threadIdx.x, wave = tid >> 5, lane = tid & 31;
    const int half = lane >> 4, n = lane & 15;
    const int wy = wave >> 2, wx = wave & 3;            // 2x4 waves -> 32x64 block tile
    __shared__ __align__(16) u16 a_lds[32 * 32];
    const u16* abase = tok + (size_t)b_ * LP * C_DIM;
    v8f acc = {};
    for (int kt = 0; kt < C_DIM / 32; ++kt) {
        int k0 = kt * 32;
        if (tid < 128) {
            int row = tid >> 2, seg = tid & 3;
            unsigned dst = (unsigned)(size_t)&a_lds[row * 32 + seg * 8];
            async_lds_b128(dst, abase + (size_t)(m0 + row) * C_DIM + k0 + seg * 8);
        }
        wait_async0();
        __syncthreads();
        Frag a, b;
#pragma unroll
        for (int v = 0; v < 8; ++v) {
            a.u[v] = *(const u32*)&a_lds[(wy * 16 + n) * 32 + a_kidx(v, half)];
            int o = nbase + wx * 16 + n;                 // B[k][n] = W_in[o][k]
            b.u[v] = *(const u32*)(win_bf + (size_t)o * C_DIM + k0 + b_kidx(v, half));
        }
        acc = wmma_bf(a, b, acc);
        __syncthreads();
    }
#pragma unroll
    for (int v = 0; v < 8; ++v) {
        int row = m0 + wy * 16 + v + 8 * half;
        int col = nbase + wx * 16 + n;
        qkv[(size_t)(b_ * LP + row) * C3 + col] = f2bf(acc[v]);
    }
}

// ---------------------------------------------------------------- kernel 3
// Flash attention per (window, head): online softmax, O = softmax(QK^T*s+mask) V.
// K staged to LDS via async-to-LDS; V staged transposed so every B-fragment
// read is a pair of ds_load_b128.
__global__ void k_attention(const u16* __restrict__ qkv, const float* __restrict__ maskadd,
                            u16* __restrict__ attno) {
    const int blk = blockIdx.x;
    const int b_ = blk / NHEAD, head = blk % NHEAD;
    const int tid = threadIdx.x, wave = tid >> 5, lane = tid & 31;
    const int half = lane >> 4, n = lane & 15;
    __shared__ __align__(16) u16 k_lds[LP * 64];      // K  [key][d]      36 KB
    __shared__ __align__(16) u16 vt_lds[64 * LP];     // V^T [d][key]     36 KB
    __shared__ u16 plds[8][16 * 32];                  // P bounce          8 KB
    const u16* base = qkv + (size_t)b_ * LP * C3;
    const float* mrow_base = maskadd + b_ * LP;

    // --- stage K via async-to-LDS (row-major copy, 128-bit granules)
    for (int i = tid; i < LP * 8; i += 256) {
        int row = i >> 3, seg = i & 7;
        unsigned dst = (unsigned)(size_t)&k_lds[row * 64 + seg * 8];
        async_lds_b128(dst, base + (size_t)row * C3 + C_DIM + head * 64 + seg * 8);
    }
    // --- stage V transposed (scalar gather once per block, shared by 8 waves)
    for (int i = tid; i < LP * 64; i += 256) {
        int key = i >> 6, dd = i & 63;
        vt_lds[dd * LP + key] = base[(size_t)key * C3 + 2 * C_DIM + head * 64 + dd];
    }
    wait_async0();
    __syncthreads();

    for (int mt = wave; mt < LP / 16; mt += 8) {
        int m0 = mt * 16;
        Frag aQ0, aQ1;
        {
            const u16* qrow = base + (size_t)(m0 + n) * C3 + head * 64;
#pragma unroll
            for (int v = 0; v < 8; ++v) {
                int k = a_kidx(v, half);
                aQ0.u[v] = *(const u32*)(qrow + k);
                aQ1.u[v] = *(const u32*)(qrow + 32 + k);
            }
        }
        float mrun[8], srun[8];
        v8f o[4];
#pragma unroll
        for (int v = 0; v < 8; ++v) { mrun[v] = -1e30f; srun[v] = 0.0f; }
#pragma unroll
        for (int c16 = 0; c16 < 4; ++c16) o[c16] = (v8f){};

        for (int kb = 0; kb < LP / 32; ++kb) {
            // --- scores S (16x32) = Q_tile (16x64) . K_block^T   (K frags from LDS)
            Frag b00, b01, b10, b11;
            {
                const u16* kr0 = &k_lds[(kb * 32 + n) * 64];
                const u16* kr1 = &k_lds[(kb * 32 + 16 + n) * 64];
#pragma unroll
                for (int v = 0; v < 8; ++v) {
                    int kd = b_kidx(v, half);
                    b00.u[v] = *(const u32*)(kr0 + kd);
                    b01.u[v] = *(const u32*)(kr0 + 32 + kd);
                    b10.u[v] = *(const u32*)(kr1 + kd);
                    b11.u[v] = *(const u32*)(kr1 + 32 + kd);
                }
            }
            v8f s0 = {}, s1 = {};
            s0 = wmma_bf(aQ0, b00, s0); s0 = wmma_bf(aQ1, b01, s0);
            s1 = wmma_bf(aQ0, b10, s1); s1 = wmma_bf(aQ1, b11, s1);
            float mk0 = mrow_base[kb * 32 + n];
            float mk1 = mrow_base[kb * 32 + 16 + n];
            float alpha[8];
#pragma unroll
            for (int v = 0; v < 8; ++v) {
                s0[v] = s0[v] * 0.125f + mk0;
                s1[v] = s1[v] * 0.125f + mk1;
                float t = fmaxf(s0[v], s1[v]);
                t = fmaxf(t, __shfl_xor(t, 1, 32));
                t = fmaxf(t, __shfl_xor(t, 2, 32));
                t = fmaxf(t, __shfl_xor(t, 4, 32));
                t = fmaxf(t, __shfl_xor(t, 8, 32));      // per-row max over 16 lanes
                float mn = fmaxf(mrun[v], t);
                alpha[v] = __expf(mrun[v] - mn);
                mrun[v] = mn;
                float p0 = __expf(s0[v] - mn);
                float p1 = __expf(s1[v] - mn);
                float rs = p0 + p1;
                rs += __shfl_xor(rs, 1, 32);
                rs += __shfl_xor(rs, 2, 32);
                rs += __shfl_xor(rs, 4, 32);
                rs += __shfl_xor(rs, 8, 32);
                srun[v] = srun[v] * alpha[v] + rs;
                int row = v + 8 * half;                   // C-layout row of this element
                plds[wave][row * 32 + n]      = f2bf(p0);
                plds[wave][row * 32 + 16 + n] = f2bf(p1);
            }
#pragma unroll
            for (int c16 = 0; c16 < 4; ++c16)
#pragma unroll
                for (int v = 0; v < 8; ++v) o[c16][v] *= alpha[v];
            wait_ds0();                                   // in-wave LDS ordering for C->A relayout
            Frag aP;
#pragma unroll
            for (int v = 0; v < 8; ++v)
                aP.u[v] = *(const u32*)&plds[wave][n * 32 + a_kidx(v, half)];
            // --- O += P (16x32) . V_block (32x64)   (V frags from transposed LDS)
#pragma unroll
            for (int c16 = 0; c16 < 4; ++c16) {
                Frag bv;
#pragma unroll
                for (int v = 0; v < 8; ++v)
                    bv.u[v] = *(const u32*)&vt_lds[(c16 * 16 + n) * LP + kb * 32 + b_kidx(v, half)];
                o[c16] = wmma_bf(aP, bv, o[c16]);
            }
        }
        // normalize + store bf16
#pragma unroll
        for (int v = 0; v < 8; ++v) {
            float inv = 1.0f / srun[v];
            int row = m0 + v + 8 * half;
#pragma unroll
            for (int c16 = 0; c16 < 4; ++c16)
                attno[(size_t)(b_ * LP + row) * C_DIM + head * 64 + c16 * 16 + n] = f2bf(o[c16][v] * inv);
        }
    }
}

// ---------------------------------------------------------------- kernel 4
// Query token mean over 25 windows (repeat-major): t[b] = mean_rr attno[rr*4+b, 196+j]
__global__ void k_qmean(const u16* __restrict__ attno, u16* __restrict__ tq) {
    int idx = blockIdx.x * 256 + threadIdx.x;
    if (idx >= 256 * C_DIM) return;
    int c = idx % C_DIM, row = idx / C_DIM;
    int b = row / 64, j = row % 64;
    float s = 0.0f;
    for (int rr = 0; rr < 25; ++rr)
        s += bf2f(attno[(size_t)((rr * 4 + b) * LP + 196 + j) * C_DIM + c]);
    tq[(size_t)row * C_DIM + c] = f2bf(s * (1.0f / 25.0f));
}

// ---------------------------------------------------------------- kernel 5
// xo = window_reverse(attn_out)[:, :H, :W] @ W_out^T + b_out  (gather fused into A staging)
__global__ void k_xo_gemm(const u16* __restrict__ attno, const u16* __restrict__ wout_bf,
                          const float* __restrict__ bias, float* __restrict__ out) {
    const int nbase = blockIdx.x * 64;
    const int m0    = blockIdx.y * 32;
    const int tid = threadIdx.x, wave = tid >> 5, lane = tid & 31;
    const int half = lane >> 4, n = lane & 15;
    const int wy = wave >> 2, wx = wave & 3;
    __shared__ __align__(16) u16 a_lds[32 * 32];
    v8f acc = {};
    for (int kt = 0; kt < C_DIM / 32; ++kt) {
        int k0 = kt * 32;
        if (tid < 128) {
            int row = tid >> 2, seg = tid & 3;
            int p = m0 + row;
            int b = p >> 12, rem = p & 4095;
            int y = rem >> 6, x = rem & 63;
            int wyy = y / 14, wxx = x / 14;
            int l  = (y - wyy * 14) * 14 + (x - wxx * 14);
            int b_ = b * 25 + wyy * 5 + wxx;
            unsigned dst = (unsigned)(size_t)&a_lds[row * 32 + seg * 8];
            async_lds_b128(dst, attno + (size_t)(b_ * LP + l) * C_DIM + k0 + seg * 8);
        }
        wait_async0();
        __syncthreads();
        Frag a, b;
#pragma unroll
        for (int v = 0; v < 8; ++v) {
            a.u[v] = *(const u32*)&a_lds[(wy * 16 + n) * 32 + a_kidx(v, half)];
            int oc = nbase + wx * 16 + n;
            b.u[v] = *(const u32*)(wout_bf + (size_t)oc * C_DIM + k0 + b_kidx(v, half));
        }
        acc = wmma_bf(a, b, acc);
        __syncthreads();
    }
    int oc = nbase + wx * 16 + n;
    float bs = bias[oc];
#pragma unroll
    for (int v = 0; v < 8; ++v) {
        int p = m0 + wy * 16 + v + 8 * half;
        out[(size_t)p * C_DIM + oc] = acc[v] + bs;
    }
}

// ---------------------------------------------------------------- kernel 6
// qo = t @ W_out^T + b_out  (small, direct fragment loads)
__global__ void k_qo_gemm(const u16* __restrict__ tq, const u16* __restrict__ wout_bf,
                          const float* __restrict__ bias, float* __restrict__ out) {
    const int nbase = blockIdx.x * 64;
    const int m0    = blockIdx.y * 32;
    const int tid = threadIdx.x, wave = tid >> 5, lane = tid & 31;
    const int half = lane >> 4, n = lane & 15;
    const int wy = wave >> 2, wx = wave & 3;
    v8f acc = {};
    for (int kt = 0; kt < C_DIM / 32; ++kt) {
        int k0 = kt * 32;
        Frag a, b;
#pragma unroll
        for (int v = 0; v < 8; ++v) {
            a.u[v] = *(const u32*)(tq + (size_t)(m0 + wy * 16 + n) * C_DIM + k0 + a_kidx(v, half));
            int oc = nbase + wx * 16 + n;
            b.u[v] = *(const u32*)(wout_bf + (size_t)oc * C_DIM + k0 + b_kidx(v, half));
        }
        acc = wmma_bf(a, b, acc);
    }
    int oc = nbase + wx * 16 + n;
    float bs = bias[oc];
#pragma unroll
    for (int v = 0; v < 8; ++v) {
        int row = m0 + wy * 16 + v + 8 * half;
        out[(size_t)row * C_DIM + oc] = acc[v] + bs;
    }
}

// ---------------------------------------------------------------- launch
extern "C" void kernel_launch(void* const* d_in, const int* in_sizes, int n_in,
                              void* d_out, int out_size, void* d_ws, size_t ws_size,
                              hipStream_t stream) {
    (void)in_sizes; (void)n_in; (void)out_size; (void)ws_size;
    const float* x     = (const float*)d_in[0];
    const float* q     = (const float*)d_in[1];
    const float* W_in  = (const float*)d_in[2];
    const float* W_out = (const float*)d_in[3];
    const float* b_out = (const float*)d_in[4];
    const int*   qmask = (const int*)d_in[5];
    float* out = (float*)d_out;

    char* ws = (char*)d_ws;
    u16*   win_bf  = (u16*)(ws + 0);                       //  2304*768  bf16
    u16*   wout_bf = (u16*)(ws + 3538944);                 //   768*768  bf16
    u16*   tok     = (u16*)(ws + 4718592);                 // 100*288*768 bf16
    float* maskadd = (float*)(ws + 48955392);              // 100*288 f32
    u16*   qkv     = (u16*)(ws + 49070592);                // 100*288*2304 bf16
    u16*   attno   = (u16*)(ws + 181780992);               // 100*288*768 bf16
    u16*   tq      = (u16*)(ws + 226017792);               // 256*768 bf16

    k_convert_weights<<<9216, 256, 0, stream>>>(W_in, W_out, win_bf, wout_bf);
    k_pack<<<86400, 256, 0, stream>>>(x, q, qmask, tok, maskadd);
    k_qkv_gemm<<<dim3(C3 / 64, LP / 32, NWIN), 256, 0, stream>>>(tok, win_bf, qkv);
    k_attention<<<NWIN * NHEAD, 256, 0, stream>>>(qkv, maskadd, attno);
    k_qmean<<<768, 256, 0, stream>>>(attno, tq);
    k_xo_gemm<<<dim3(C_DIM / 64, 16384 / 32), 256, 0, stream>>>(attno, wout_bf, b_out, out);
    k_qo_gemm<<<dim3(C_DIM / 64, 256 / 32), 256, 0, stream>>>(tq, wout_bf, b_out, out + (size_t)4 * 4096 * C_DIM);
}
